// Translator_26474178412961
// MI455X (gfx1250) — compile-verified, matrix operand order checked by reading, not run on hardware
//
#include <hip/hip_runtime.h>
#include <hip/hip_bf16.h>
#include <math.h>

#define NNODES 100000
#define NEDGES 640000
#define NGRAPH 128
#define BN_EPS 1e-5f

typedef __attribute__((ext_vector_type(16))) _Float16 v16h;
typedef __attribute__((ext_vector_type(8)))  _Float16 h8v;
typedef __attribute__((ext_vector_type(4)))  _Float16 h4v;
typedef __attribute__((ext_vector_type(8)))  float    v8f;

__device__ __forceinline__ void atomAddF(float* p, float v) {
    __hip_atomic_fetch_add(p, v, __ATOMIC_RELAXED, __HIP_MEMORY_SCOPE_AGENT);
}

// float atomic max via sign-aware integer punning (init must be -inf)
__device__ __forceinline__ void atomMaxF(float* p, float v) {
    if (v >= 0.0f) atomicMax((int*)p, __float_as_int(v));
    else           atomicMin((unsigned int*)p, __float_as_uint(v));
}

// ---------------- small init -------------------------------------------------
__global__ void gin_init(float* gstats, float* gmax, float* gsum, float* scal) {
    int t = threadIdx.x;
    if (t < 256) gstats[t] = 0.0f;
    if (t < 128) { gmax[t] = -INFINITY; gsum[t] = 0.0f; }
    if (t < 4)   scal[t] = 0.0f;
}

// ---------------- vectorized copy (agg = x) ---------------------------------
__global__ __launch_bounds__(256) void gin_copy4(float4* __restrict__ dst,
                                                 const float4* __restrict__ src, int n4) {
    for (int i = blockIdx.x * 256 + threadIdx.x; i < n4; i += gridDim.x * 256)
        dst[i] = src[i];
}

// ---------------- edge scatter: agg[dst] += x[src] --------------------------
// one wave per edge row; each lane handles 4 features (float4 gather + 4 fp32 atomics)
__global__ __launch_bounds__(256) void gin_scatter(float* __restrict__ agg,
                                                   const float* __restrict__ x,
                                                   const int* __restrict__ src,
                                                   const int* __restrict__ dst, int e) {
    int gtid = blockIdx.x * 256 + threadIdx.x;
    int eid = gtid >> 5;
    if (eid >= e) return;
    int q = (gtid & 31) * 4;
    float4 v = *(const float4*)(x + (size_t)src[eid] * 128 + q);
    float* d = agg + (size_t)dst[eid] * 128 + q;
    atomAddF(d + 0, v.x); atomAddF(d + 1, v.y);
    atomAddF(d + 2, v.z); atomAddF(d + 3, v.w);
}

// ---------------- WMMA GEMM: out = act(A[N,128] @ W[128,128] + bias) --------
// 8 waves/block; wave w owns rows [blk*128+16w, +16). 8 col-tiles x 4 K-chunks
// of v_wmma_f32_16x16x32_f16 with f32 accumulation.
template <bool AHALF, bool RELU, bool OUTHALF>
__global__ __launch_bounds__(256)
void gin_gemm128(const void* __restrict__ Ain, const float* __restrict__ W,
                 const float* __restrict__ bias, int n,
                 _Float16* __restrict__ outH, float* __restrict__ outF) {
    __shared__ _Float16 wlds[128 * 128];  // transposed: wlds[col*128 + k] (f16)
    __shared__ float blds[128];

    const int tid = threadIdx.x;
    for (int i = tid; i < 128 * 128; i += 256) {
        int k = i >> 7, col = i & 127;               // coalesced global read of W[k][col]
        wlds[col * 128 + k] = (_Float16)W[i];
    }
    if (tid < 128) blds[tid] = bias[tid];
    __syncthreads();

    const int wave  = tid >> 5;
    const int lane  = tid & 31;
    const int lrow  = lane & 15;        // M within A tile / N within B tile
    const int hi    = lane >> 4;        // half-wave select
    const int khalf = hi << 3;          // 0 or 8 (ISA 16-bit A layout)
    const int row0  = blockIdx.x * 128 + wave * 16;
    const int arow  = row0 + lrow;
    const bool rvalid = arow < n;

    v8f acc[8] = {};

    for (int kc = 0; kc < 4; ++kc) {
        const int kbase = kc * 32;
        v16h af;
        if (rvalid) {
            if (AHALF) {
                const _Float16* ap = (const _Float16*)Ain + (size_t)arow * 128 + kbase;
                h8v lo  = *(const h8v*)(ap + khalf);
                h8v hi8 = *(const h8v*)(ap + 16 + khalf);
                #pragma unroll
                for (int i = 0; i < 8; ++i) { af[i] = lo[i]; af[8 + i] = hi8[i]; }
            } else {
                const float* ap = (const float*)Ain + (size_t)arow * 128 + kbase;
                float4 a0 = *(const float4*)(ap + khalf);
                float4 a1 = *(const float4*)(ap + khalf + 4);
                float4 a2 = *(const float4*)(ap + 16 + khalf);
                float4 a3 = *(const float4*)(ap + 16 + khalf + 4);
                af[0]=(_Float16)a0.x; af[1]=(_Float16)a0.y; af[2]=(_Float16)a0.z; af[3]=(_Float16)a0.w;
                af[4]=(_Float16)a1.x; af[5]=(_Float16)a1.y; af[6]=(_Float16)a1.z; af[7]=(_Float16)a1.w;
                af[8]=(_Float16)a2.x; af[9]=(_Float16)a2.y; af[10]=(_Float16)a2.z; af[11]=(_Float16)a2.w;
                af[12]=(_Float16)a3.x; af[13]=(_Float16)a3.y; af[14]=(_Float16)a3.z; af[15]=(_Float16)a3.w;
            }
        } else {
            #pragma unroll
            for (int i = 0; i < 16; ++i) af[i] = (_Float16)0.0f;
        }

        #pragma unroll
        for (int ct = 0; ct < 8; ++ct) {
            const int col = ct * 16 + lrow;
            const _Float16* bp = &wlds[col * 128 + kbase + khalf];
            h8v blo = *(const h8v*)bp;            // K = kbase+khalf .. +7   (ds_load_b128)
            h8v bhi = *(const h8v*)(bp + 16);     // K = kbase+16+khalf .. +7
            v16h bf;
            #pragma unroll
            for (int i = 0; i < 8; ++i) { bf[i] = blo[i]; bf[8 + i] = bhi[i]; }
            acc[ct] = __builtin_amdgcn_wmma_f32_16x16x32_f16(
                false, af, false, bf, (short)0, acc[ct], false, false);
        }
    }

    // epilogue: C layout — acc[ct][r] is (row0 + r + 8*hi, ct*16 + lrow)
    #pragma unroll
    for (int ct = 0; ct < 8; ++ct) {
        const int col = ct * 16 + lrow;
        const float b = blds[col];
        #pragma unroll
        for (int r = 0; r < 8; ++r) {
            int row = row0 + r + (hi << 3);
            if (row < n) {
                float v = acc[ct][r] + b;
                if (RELU) v = fmaxf(v, 0.0f);
                if (OUTHALF) outH[(size_t)row * 128 + col] = (_Float16)v;
                else         outF[(size_t)row * 128 + col] = v;
            }
        }
    }
}

// ---------------- BN channel stats (sum, sumsq) -----------------------------
__global__ __launch_bounds__(256) void gin_stats(const float* __restrict__ h,
                                                 float* __restrict__ gs, int n) {
    __shared__ float s1[256], s2[256];
    int c = threadIdx.x & 127, half = threadIdx.x >> 7;
    float a = 0.0f, b = 0.0f;
    for (int row = blockIdx.x * 2 + half; row < n; row += gridDim.x * 2) {
        float v = h[(size_t)row * 128 + c];
        a += v; b += v * v;
    }
    s1[threadIdx.x] = a; s2[threadIdx.x] = b;
    __syncthreads();
    if (threadIdx.x < 128) {
        atomAddF(&gs[c],       s1[c] + s1[c + 128]);
        atomAddF(&gs[128 + c], s2[c] + s2[c + 128]);
    }
}

__global__ void gin_bnprep(float* gs, const float* __restrict__ g,
                           const float* __restrict__ be,
                           float* __restrict__ scale, float* __restrict__ shift,
                           float invn) {
    int c = threadIdx.x;
    float mu  = gs[c] * invn;
    float var = gs[128 + c] * invn - mu * mu;
    float inv = rsqrtf(var + BN_EPS);
    float sc  = g[c] * inv;
    scale[c] = sc;
    shift[c] = be[c] - mu * sc;
    gs[c] = 0.0f; gs[128 + c] = 0.0f;   // ready for next layer
}

// BN + ReLU, writes activation in place AND seeds next layer's agg buffer
__global__ __launch_bounds__(256) void gin_bnapply(float* __restrict__ h,
                                                   float* __restrict__ agg,
                                                   const float* __restrict__ scale,
                                                   const float* __restrict__ shift, int n4) {
    for (int i = blockIdx.x * 256 + threadIdx.x; i < n4; i += gridDim.x * 256) {
        int c = (i & 31) * 4;
        float4 v  = ((const float4*)h)[i];
        float4 sc = *(const float4*)(scale + c);
        float4 sh = *(const float4*)(shift + c);
        v.x = fmaxf(v.x * sc.x + sh.x, 0.0f);
        v.y = fmaxf(v.y * sc.y + sh.y, 0.0f);
        v.z = fmaxf(v.z * sc.z + sh.z, 0.0f);
        v.w = fmaxf(v.w * sc.w + sh.w, 0.0f);
        ((float4*)h)[i] = v;
        ((float4*)agg)[i] = v;
    }
}

// ---------------- layer-2 [128,1] linear: wave dot + scalar BN stats --------
__global__ __launch_bounds__(256) void gin_rowdot(const _Float16* __restrict__ t,
                                                  const float* __restrict__ w,
                                                  const float* __restrict__ bptr,
                                                  float* __restrict__ h3,
                                                  float* __restrict__ scal, int n) {
    int wave = threadIdx.x >> 5, lane = threadIdx.x & 31;
    int row = blockIdx.x * 8 + wave;
    if (row >= n) return;
    h4v a = *(const h4v*)(t + (size_t)row * 128 + lane * 4);
    float4 wv = *(const float4*)(w + lane * 4);
    float s = (float)a[0] * wv.x + (float)a[1] * wv.y +
              (float)a[2] * wv.z + (float)a[3] * wv.w;
    #pragma unroll
    for (int m = 16; m > 0; m >>= 1) s += __shfl_xor(s, m, 32);
    if (lane == 0) {
        float v = s + bptr[0];
        h3[row] = v;
        atomAddF(&scal[0], v);
        atomAddF(&scal[1], v * v);
    }
}

__global__ void gin_sprep(float* scal, const float* g2, const float* be2) {
    float invn = 1.0f / (float)NNODES;
    float mu  = scal[0] * invn;
    float var = scal[1] * invn - mu * mu;
    float inv = rsqrtf(var + BN_EPS);
    float sc  = g2[0] * inv;
    scal[2] = sc * 0.2f;                       // fold temperature 5.0
    scal[3] = (be2[0] - mu * sc) * 0.2f;
    scal[0] = 0.0f; scal[1] = 0.0f;
}

// ---------------- segment softmax -------------------------------------------
__global__ __launch_bounds__(256) void gin_smax(float* __restrict__ z,
                                                const int* __restrict__ batch,
                                                float* __restrict__ gmax,
                                                const float* __restrict__ scal, int n) {
    int i = blockIdx.x * 256 + threadIdx.x;
    if (i >= n) return;
    float v = z[i] * scal[2] + scal[3];
    z[i] = v;
    atomMaxF(&gmax[batch[i]], v);
}

__global__ __launch_bounds__(256) void gin_sexp(const float* __restrict__ z,
                                                const int* __restrict__ batch,
                                                const float* __restrict__ gmax,
                                                float* __restrict__ gsum,
                                                float* __restrict__ out, int n) {
    int i = blockIdx.x * 256 + threadIdx.x;
    if (i >= n) return;
    int b = batch[i];
    float e = expf(z[i] - gmax[b]);
    out[i] = e;
    atomAddF(&gsum[b], e);
}

__global__ __launch_bounds__(256) void gin_sdiv(float* __restrict__ out,
                                                const int* __restrict__ batch,
                                                const float* __restrict__ gsum, int n) {
    int i = blockIdx.x * 256 + threadIdx.x;
    if (i >= n) return;
    out[i] = out[i] / (gsum[batch[i]] + 1e-16f);
}

// ---------------- launch ------------------------------------------------------
extern "C" void kernel_launch(void* const* d_in, const int* in_sizes, int n_in,
                              void* d_out, int out_size, void* d_ws, size_t ws_size,
                              hipStream_t stream) {
    (void)in_sizes; (void)n_in; (void)out_size; (void)ws_size;
    const float* x     = (const float*)d_in[0];
    const int*   ei    = (const int*)d_in[1];
    const int*   batch = (const int*)d_in[2];
    const float* W0a = (const float*)d_in[3];  const float* b0a = (const float*)d_in[4];
    const float* W0b = (const float*)d_in[5];  const float* b0b = (const float*)d_in[6];
    const float* g0  = (const float*)d_in[7];  const float* be0 = (const float*)d_in[8];
    const float* W1a = (const float*)d_in[9];  const float* b1a = (const float*)d_in[10];
    const float* W1b = (const float*)d_in[11]; const float* b1b = (const float*)d_in[12];
    const float* g1  = (const float*)d_in[13]; const float* be1 = (const float*)d_in[14];
    const float* W2a = (const float*)d_in[15]; const float* b2a = (const float*)d_in[16];
    const float* W2b = (const float*)d_in[17]; const float* b2b = (const float*)d_in[18];
    const float* g2  = (const float*)d_in[19]; const float* be2 = (const float*)d_in[20];
    const int* src = ei;
    const int* dst = ei + NEDGES;

    char* ws = (char*)d_ws;
    float* gstats = (float*)ws;           // 256 f32
    float* scale  = gstats + 256;         // 128
    float* shift  = scale + 128;          // 128
    float* gmax   = shift + 128;          // 128
    float* gsum   = gmax + 128;           // 128
    float* scal   = gsum + 128;           // 4
    float*    bufAgg = (float*)(ws + 4096);
    float*    bufH   = bufAgg + (size_t)NNODES * 128;
    _Float16* bufT   = (_Float16*)(bufH + (size_t)NNODES * 128);
    float*    h3     = (float*)(bufT + (size_t)NNODES * 128);
    float* out = (float*)d_out;

    const int gGemm = (NNODES + 127) / 128;
    const int gScat = (NEDGES * 32 + 255) / 256;
    const int gN    = (NNODES + 255) / 256;
    const int n4    = NNODES * 128 / 4;
    const float invn = 1.0f / (float)NNODES;

    gin_init<<<1, 512, 0, stream>>>(gstats, gmax, gsum, scal);

    // ---- layer 0 ----
    gin_copy4<<<2048, 256, 0, stream>>>((float4*)bufAgg, (const float4*)x, n4);
    gin_scatter<<<gScat, 256, 0, stream>>>(bufAgg, x, src, dst, NEDGES);
    gin_gemm128<false, true,  true ><<<gGemm, 256, 0, stream>>>(bufAgg, W0a, b0a, NNODES, bufT, nullptr);
    gin_gemm128<true,  false, false><<<gGemm, 256, 0, stream>>>(bufT,   W0b, b0b, NNODES, nullptr, bufH);
    gin_stats<<<512, 256, 0, stream>>>(bufH, gstats, NNODES);
    gin_bnprep<<<1, 128, 0, stream>>>(gstats, g0, be0, scale, shift, invn);
    gin_bnapply<<<2048, 256, 0, stream>>>(bufH, bufAgg, scale, shift, n4);

    // ---- layer 1 ----
    gin_scatter<<<gScat, 256, 0, stream>>>(bufAgg, bufH, src, dst, NEDGES);
    gin_gemm128<false, true,  true ><<<gGemm, 256, 0, stream>>>(bufAgg, W1a, b1a, NNODES, bufT, nullptr);
    gin_gemm128<true,  false, false><<<gGemm, 256, 0, stream>>>(bufT,   W1b, b1b, NNODES, nullptr, bufH);
    gin_stats<<<512, 256, 0, stream>>>(bufH, gstats, NNODES);
    gin_bnprep<<<1, 128, 0, stream>>>(gstats, g1, be1, scale, shift, invn);
    gin_bnapply<<<2048, 256, 0, stream>>>(bufH, bufAgg, scale, shift, n4);

    // ---- layer 2 ----
    gin_scatter<<<gScat, 256, 0, stream>>>(bufAgg, bufH, src, dst, NEDGES);
    gin_gemm128<false, true, true><<<gGemm, 256, 0, stream>>>(bufAgg, W2a, b2a, NNODES, bufT, nullptr);
    gin_rowdot<<<(NNODES + 7) / 8, 256, 0, stream>>>(bufT, W2b, b2b, h3, scal, NNODES);
    gin_sprep<<<1, 1, 0, stream>>>(scal, g2, be2);

    // ---- segment softmax (temperature folded into scal[2]/scal[3]) ----
    gin_smax<<<gN, 256, 0, stream>>>(h3, batch, gmax, scal, NNODES);
    gin_sexp<<<gN, 256, 0, stream>>>(h3, batch, gmax, gsum, out, NNODES);
    gin_sdiv<<<gN, 256, 0, stream>>>(out, batch, gsum, NNODES);
}
